// STMoE_DistilGPT2_old_28346784154228
// MI455X (gfx1250) — matrix operand order repaired and checked by reading, not compiled, not run on hardware
//
#include <hip/hip_runtime.h>
#include <hip/hip_bf16.h>

typedef __attribute__((ext_vector_type(16))) __bf16 v16bf;
typedef __attribute__((ext_vector_type(8)))  float  v8f;

#define DIM   768
#define NHEAD 12
#define HDIM  64
#define NLAY  6
#define VOC   50258
#define FFD   3072
#define NE    8
#define MOEH  2048
#define CAP   80
#define SEQ   512
#define BB    2
#define NTOK  (BB*SEQ)

// pack two bf16 into one 32-bit word
union bfpair { __bf16 h[2]; unsigned u; };

// ---------------------------------------------------------------------------
// Stage one 128x32 A tile and one 32x128 B tile (fp32 -> bf16) into LDS in
// WMMA fragment order:
//   A frag: lane = (m&15) + 16*((k>>3)&1), slot = (k&7) + 8*(k>>4)
//   B frag: lane = (n&15) + 16*((k>>3)&1), slot = (k&7) + 8*(k>>4)
// K must be a multiple of 32 (true for every call site); M/N edges guarded by
// clamped addresses + selects (no exec branching).
// Phase 1 issues ALL 16 global b64 loads into registers (single wait region),
// phase 2 converts + ds_stores -- avoids per-load full-latency stalls.
// ---------------------------------------------------------------------------
__device__ __forceinline__ void stage_tiles(
    __bf16 (*As)[32][16], __bf16 (*Bs)[32][16],
    const float* __restrict__ Ab, const float* __restrict__ Bb,
    int m0, int n0, int k0, int M, int N, int lda, int ldb, int transB, int tid)
{
    float a0[8], a1[8], b0[8], b1[8];

    // ---- phase 1: gather (16x global_load_b64 in flight) ----
#pragma unroll
    for (int it = 0; it < 8; ++it) {
        int pidx = tid + it * 256;          // 2048 pairs: 128 rows x 16 kpairs
        int r = pidx >> 4;
        int k = (pidx & 15) * 2;
        int gm = m0 + r;
        int gmc = gm < M ? gm : M - 1;      // clamp, no branch
        const float2 v = *(const float2*)(Ab + (long)gmc * lda + k0 + k);
        a0[it] = v.x; a1[it] = v.y;
    }
#pragma unroll
    for (int it = 0; it < 8; ++it) {
        int pidx = tid + it * 256;          // 2048 pairs: 16 kpairs x 128 n
        int k = (pidx >> 7) * 2;
        int nn = pidx & 127;
        int gn = n0 + nn;
        int gnc = gn < N ? gn : N - 1;      // clamp, no branch
        if (transB) {
            const float2 v = *(const float2*)(Bb + (long)gnc * ldb + k0 + k);
            b0[it] = v.x; b1[it] = v.y;
        } else {
            b0[it] = Bb[(long)(k0 + k) * ldb + gnc];
            b1[it] = Bb[(long)(k0 + k + 1) * ldb + gnc];
        }
    }

    // ---- phase 2: convert + store to LDS in fragment order ----
#pragma unroll
    for (int it = 0; it < 8; ++it) {
        int pidx = tid + it * 256;
        int r = pidx >> 4;
        int k = (pidx & 15) * 2;
        bool ok = (m0 + r) < M;
        bfpair p;
        p.h[0] = (__bf16)(ok ? a0[it] : 0.f);
        p.h[1] = (__bf16)(ok ? a1[it] : 0.f);
        int dl = (r & 15) + 16 * ((k >> 3) & 1);
        int dsl = (k & 7) + 8 * (k >> 4);
        *(unsigned*)&As[r >> 4][dl][dsl] = p.u;
    }
#pragma unroll
    for (int it = 0; it < 8; ++it) {
        int pidx = tid + it * 256;
        int k = (pidx >> 7) * 2;
        int nn = pidx & 127;
        bool ok = (n0 + nn) < N;
        bfpair p;
        p.h[0] = (__bf16)(ok ? b0[it] : 0.f);
        p.h[1] = (__bf16)(ok ? b1[it] : 0.f);
        int dl = (nn & 15) + 16 * ((k >> 3) & 1);
        int dsl = (k & 7) + 8 * (k >> 4);
        *(unsigned*)&Bs[nn >> 4][dl][dsl] = p.u;
    }
}

// ---------------------------------------------------------------------------
// Generic batched GEMM:  C = act(alpha * A@B(+bias)) (+res)
// A: (M,K) row-major fp32, B: (K,N) row-major (or (N,K) if transB) fp32.
// Macro tile 128(M) x 128(N), 256 threads = 8 waves (wave32).
// Wave (wrow 0..3, wcol 0..1) owns a 32x64 strip: two 16-row A fragments that
// share each B fragment -> 8 v_wmma per K-step for 12 ds_load_b128.
// LDS tiles are double-buffered: next K-step is staged while WMMAs consume the
// current one; a single barrier per K-step.
// ---------------------------------------------------------------------------
__global__ __launch_bounds__(256)
void gemm_bf16_wmma(const float* __restrict__ A, const float* __restrict__ B,
                    const float* bias, const float* res, float* C,
                    int M, int N, int K, int lda, int ldb, int ldc,
                    int nb2, long sA1, long sA2, long sB1, long sB2,
                    long sC1, long sC2, long sBias,
                    float alpha, int transB, int act)
{
    __shared__ __align__(32) __bf16 As2[2][8][32][16];   // 2 bufs x 128x32
    __shared__ __align__(32) __bf16 Bs2[2][8][32][16];   // 2 bufs x 32x128

    const int tid  = threadIdx.x;
    const int lane = tid & 31;
    const int w    = tid >> 5;
    const int wrow = w >> 1;       // 0..3
    const int wcol = w & 1;        // 0..1
    const int hfl  = lane >> 4;    // half-wave 0/1
    const int l16  = lane & 15;

    const int z  = blockIdx.z;
    const int b1 = z / nb2, b2 = z % nb2;
    const float* Ab = A + b1 * sA1 + b2 * sA2;
    const float* Bb = B + b1 * sB1 + b2 * sB2;
    float*       Cb = C + b1 * sC1 + b2 * sC2;
    const float* Rb = res  ? res  + b1 * sC1 + b2 * sC2 : nullptr;
    const float* Gb = bias ? bias + b1 * sBias          : nullptr;

    const int m0 = blockIdx.y * 128;
    const int n0 = blockIdx.x * 128;

    v8f acc[2][4];
#pragma unroll
    for (int ms = 0; ms < 2; ++ms)
#pragma unroll
        for (int t = 0; t < 4; ++t)
#pragma unroll
            for (int r = 0; r < 8; ++r) acc[ms][t][r] = 0.f;

    stage_tiles(As2[0], Bs2[0], Ab, Bb, m0, n0, 0, M, N, lda, ldb, transB, tid);

    int pb = 0;
    for (int k0 = 0; k0 < K; k0 += 32) {
        __syncthreads();
        // stage next tile into the other buffer (overlaps with WMMA below)
        if (k0 + 32 < K)
            stage_tiles(As2[1 - pb], Bs2[1 - pb], Ab, Bb, m0, n0, k0 + 32,
                        M, N, lda, ldb, transB, tid);
        // prefetch two tiles ahead into cache
        int pk = k0 + 64;
        if (pk < K) {
            if (tid < 128) {
                int gm = m0 + tid;
                if (gm < M) __builtin_prefetch(&Ab[(long)gm * lda + pk], 0, 1);
            } else {
                int u = tid - 128;
                if (transB) {
                    int gn = n0 + u;
                    if (gn < N) __builtin_prefetch(&Bb[(long)gn * ldb + pk], 0, 1);
                } else {
                    int gk = pk + (u >> 2), gn = n0 + (u & 3) * 32;
                    if (gn < N) __builtin_prefetch(&Bb[(long)gk * ldb + gn], 0, 1);
                }
            }
        }

        // operand fragments: one 32B vector load per fragment (2x ds_load_b128)
        const v16bf af0 = *(const v16bf*)&As2[pb][wrow * 2 + 0][lane][0];
        const v16bf af1 = *(const v16bf*)&As2[pb][wrow * 2 + 1][lane][0];
#pragma unroll
        for (int t = 0; t < 4; ++t) {
            const v16bf bv = *(const v16bf*)&Bs2[pb][wcol * 4 + t][lane][0];
            acc[0][t] = __builtin_amdgcn_wmma_f32_16x16x32_bf16(
                false, af0, false, bv, (short)0, acc[0][t], false, false);
            acc[1][t] = __builtin_amdgcn_wmma_f32_16x16x32_bf16(
                false, af1, false, bv, (short)0, acc[1][t], false, false);
        }
        pb ^= 1;
    }

    // epilogue: C/D layout -> vgpr r holds M = r + 8*h, N = lane&15
#pragma unroll
    for (int ms = 0; ms < 2; ++ms) {
#pragma unroll
        for (int t = 0; t < 4; ++t) {
#pragma unroll
            for (int r = 0; r < 8; ++r) {
                int m = m0 + (wrow * 2 + ms) * 16 + r + 8 * hfl;
                int n = n0 + wcol * 64 + t * 16 + l16;
                if (m < M && n < N) {
                    float v = alpha * acc[ms][t][r];
                    if (Gb) v += Gb[n];
                    if (act == 1) {             // gelu_new (tanh approx)
                        float x = v;
                        v = 0.5f * x * (1.f + tanhf(0.7978845608028654f *
                                                    (x + 0.044715f * x * x * x)));
                    }
                    if (Rb) v += Rb[(long)m * ldc + n];
                    Cb[(long)m * ldc + n] = v;
                }
            }
        }
    }
}

// ---------------------------------------------------------------------------
// Elementwise / reduction kernels
// ---------------------------------------------------------------------------
__global__ void embed_k(const int* __restrict__ ids, const float* __restrict__ wte,
                        const float* __restrict__ wpe, float* __restrict__ x)
{
    long i = (long)blockIdx.x * 256 + threadIdx.x;
    if (i >= (long)NTOK * DIM) return;
    int tok = (int)(i / DIM), j = (int)(i % DIM);
    int s = tok % SEQ;
    x[i] = wte[(long)ids[tok] * DIM + j] + wpe[(long)s * DIM + j];
}

__global__ __launch_bounds__(256)
void layernorm_k(const float* __restrict__ x, const float* __restrict__ g,
                 const float* __restrict__ b, float* __restrict__ y)
{
    __shared__ float red[256];
    const int row = blockIdx.x;
    const float* xr = x + (long)row * DIM;
    float s = 0.f;
    for (int j = threadIdx.x; j < DIM; j += 256) s += xr[j];
    red[threadIdx.x] = s; __syncthreads();
    for (int o = 128; o > 0; o >>= 1) { if (threadIdx.x < o) red[threadIdx.x] += red[threadIdx.x + o]; __syncthreads(); }
    float mean = red[0] / DIM; __syncthreads();
    s = 0.f;
    for (int j = threadIdx.x; j < DIM; j += 256) { float d = xr[j] - mean; s += d * d; }
    red[threadIdx.x] = s; __syncthreads();
    for (int o = 128; o > 0; o >>= 1) { if (threadIdx.x < o) red[threadIdx.x] += red[threadIdx.x + o]; __syncthreads(); }
    float inv = rsqrtf(red[0] / DIM + 1e-5f);
    for (int j = threadIdx.x; j < DIM; j += 256)
        y[(long)row * DIM + j] = (xr[j] - mean) * inv * g[j] + b[j];
}

__global__ __launch_bounds__(256)
void rmsnorm_k(const float* __restrict__ x, const float* __restrict__ g,
               float* __restrict__ y)
{
    __shared__ float red[256];
    const int row = blockIdx.x;
    const float* xr = x + (long)row * DIM;
    float s = 0.f;
    for (int j = threadIdx.x; j < DIM; j += 256) s += xr[j] * xr[j];
    red[threadIdx.x] = s; __syncthreads();
    for (int o = 128; o > 0; o >>= 1) { if (threadIdx.x < o) red[threadIdx.x] += red[threadIdx.x + o]; __syncthreads(); }
    float n = sqrtf(red[0]);
    float scale = 27.712812921102035f / fmaxf(n, 1e-12f);   // sqrt(768)/||x||
    for (int j = threadIdx.x; j < DIM; j += 256)
        y[(long)row * DIM + j] = xr[j] * scale * g[j];
}

__global__ __launch_bounds__(256)
void softmax_causal_k(float* __restrict__ att)   // rows = B*NH*S, len S
{
    __shared__ float red[256];
    const int row = blockIdx.x;
    const int q = row % SEQ;
    float* a = att + (long)row * SEQ;
    float mx = -3.0e38f;
    for (int j = threadIdx.x; j < SEQ; j += 256)
        mx = fmaxf(mx, (j <= q) ? a[j] : -3.0e38f);
    red[threadIdx.x] = mx; __syncthreads();
    for (int o = 128; o > 0; o >>= 1) { if (threadIdx.x < o) red[threadIdx.x] = fmaxf(red[threadIdx.x], red[threadIdx.x + o]); __syncthreads(); }
    mx = red[0]; __syncthreads();
    float s = 0.f;
    for (int j = threadIdx.x; j < SEQ; j += 256) {
        float e = (j <= q) ? __expf(a[j] - mx) : 0.f;
        a[j] = e; s += e;
    }
    red[threadIdx.x] = s; __syncthreads();
    for (int o = 128; o > 0; o >>= 1) { if (threadIdx.x < o) red[threadIdx.x] += red[threadIdx.x + o]; __syncthreads(); }
    float inv = 1.f / red[0];
    for (int j = threadIdx.x; j < SEQ; j += 256) a[j] *= inv;
}

__global__ __launch_bounds__(256)
void geglu_k(const float* __restrict__ t, const float* __restrict__ mb,
             float* __restrict__ y, int H, int rowsPerGroup)
{
    const int row = blockIdx.x;
    const int grp = row / rowsPerGroup;
    const float* tr  = t  + (long)row * 2 * H;
    const float* mbg = mb + (long)grp * H;
    for (int j = threadIdx.x; j < H; j += 256) {
        float a = tr[j], gate = tr[H + j];
        float ge = 0.5f * gate * (1.f + erff(gate * 0.7071067811865476f));
        y[(long)row * H + j] = a * ge * mbg[j];
    }
}

__global__ void add_k(float* __restrict__ x, const float* __restrict__ y, long n)
{
    long i = (long)blockIdx.x * 256 + threadIdx.x;
    if (i < n) x[i] += y[i];
}

__global__ void zero_k(float* __restrict__ p, long n)
{
    long i = (long)blockIdx.x * 256 + threadIdx.x;
    if (i < n) p[i] = 0.f;
}

// ---------------------------------------------------------------------------
// MoE gating
// ---------------------------------------------------------------------------
__device__ __forceinline__ float hash_u01(unsigned seed, unsigned key)
{
    unsigned x = seed * 0x9E3779B1u ^ key * 0x85EBCA77u;
    x ^= x >> 16; x *= 0x7FEB352Du; x ^= x >> 15; x *= 0x846CA68Bu; x ^= x >> 16;
    return (float)(x >> 8) * (1.0f / 16777216.0f);
}

__global__ __launch_bounds__(64)
void gate_k(const float* __restrict__ h, const float* __restrict__ gw,
            int* __restrict__ tok_e, float* __restrict__ tok_g,
            float* __restrict__ rawsum, float* __restrict__ zsum)
{
    const int tok = blockIdx.x;                 // 0..NTOK-1
    const int b = tok / SEQ;
    __shared__ float lg[NE];
    if (threadIdx.x < NE) {
        float s = 0.f;
        const float* hr = h + (long)tok * DIM;
        for (int j = 0; j < DIM; ++j) s += hr[j] * gw[(long)j * NE + threadIdx.x];
        lg[threadIdx.x] = s;
    }
    __syncthreads();
    if (threadIdx.x == 0) {
        float mx = lg[0];
        for (int i = 1; i < NE; ++i) mx = fmaxf(mx, lg[i]);
        float p[NE], ssum = 0.f;
        for (int i = 0; i < NE; ++i) { p[i] = __expf(lg[i] - mx); ssum += p[i]; }
        float lse = mx + __logf(ssum);
        for (int i = 0; i < NE; ++i) p[i] /= ssum;
        int i0 = 0;
        for (int i = 1; i < NE; ++i) if (p[i] > p[i0]) i0 = i;
        int i1 = -1;
        for (int i = 0; i < NE; ++i) { if (i == i0) continue; if (i1 < 0 || p[i] > p[i1]) i1 = i; }
        float g0 = p[i0], g1 = p[i1];
        float denom = fmaxf(g0 + g1, 1e-9f);
        tok_e[tok * 2] = i0; tok_e[tok * 2 + 1] = i1;
        tok_g[tok * 2] = g0 / denom; tok_g[tok * 2 + 1] = g1 / denom;
        for (int i = 0; i < NE; ++i) atomicAdd(&rawsum[b * NE + i], p[i]);
        atomicAdd(zsum, lse * lse);
    }
}

// one block; threads 0..BB-1 do the serial per-batch capacity scan
__global__ __launch_bounds__(32)
void moe_scan_k(const int* __restrict__ tok_e, const float* __restrict__ tok_g,
                int* __restrict__ asg_e, int* __restrict__ asg_c,
                float* __restrict__ asg_g,
                const float* __restrict__ rawsum, const float* __restrict__ zsum,
                float* __restrict__ auxacc, unsigned seed)
{
    __shared__ float balsh[BB];
    const int b = threadIdx.x;
    if (b < BB) {
        int c0[NE], c1[NE], prev[NE], cnt1[NE];
        for (int e = 0; e < NE; ++e) { c0[e] = 0; c1[e] = 0; cnt1[e] = 0; }
        // pass 0: top-1 always routed; position = exclusive cumsum (uncapped)
        for (int n = 0; n < SEQ; ++n) {
            int tok = b * SEQ + n;
            int e0 = tok_e[tok * 2];
            cnt1[e0]++;
            int pos = c0[e0]++;
            int ai = (0 * BB + b) * SEQ + n;
            asg_e[ai] = e0; asg_c[ai] = pos;
            asg_g[ai] = (pos < CAP) ? tok_g[tok * 2] : 0.f;
        }
        for (int e = 0; e < NE; ++e) prev[e] = (c0[e] < CAP) ? c0[e] : CAP;
        // pass 1: top-2 with stochastic routing (probs < gate/THRESH)
        for (int n = 0; n < SEQ; ++n) {
            int tok = b * SEQ + n;
            int e1 = tok_e[tok * 2 + 1];
            float g1 = tok_g[tok * 2 + 1];
            float u = hash_u01(seed, (unsigned)(b * SEQ + n));
            int ai = (1 * BB + b) * SEQ + n;
            if (u < g1 * 5.0f) {                 // g1 / 0.2
                int pos = c1[e1]++ + prev[e1];
                asg_e[ai] = e1; asg_c[ai] = pos;
                asg_g[ai] = (pos < CAP) ? g1 : 0.f;
            } else {
                asg_e[ai] = e1; asg_c[ai] = 0; asg_g[ai] = 0.f;
            }
        }
        float bal = 0.f;
        for (int e = 0; e < NE; ++e)
            bal += (rawsum[b * NE + e] / SEQ) * ((float)cnt1[e] / SEQ);
        balsh[b] = bal;
    }
    __syncthreads();
    if (threadIdx.x == 0) {
        float balance = (balsh[0] + balsh[1]) / (float)(BB * NE) * (float)(NE * NE);
        float zl = zsum[0] / (float)(BB * SEQ);
        atomicAdd(auxacc, 0.01f * balance + 0.001f * zl);
    }
}

__global__ __launch_bounds__(256)
void moe_scatter_k(const float* __restrict__ h, const int* __restrict__ asg_e,
                   const int* __restrict__ asg_c, const float* __restrict__ asg_g,
                   float* __restrict__ ein)
{
    const int ai = blockIdx.x;                  // (k,b,n) flattened: 2*NTOK
    if (asg_g[ai] <= 0.f) return;               // dropped or unrouted
    const int tok = ai % NTOK;
    const int b = tok / SEQ;
    const int e = asg_e[ai], c = asg_c[ai];
    const long dst = ((long)e * (BB * CAP) + b * CAP + c) * DIM;
    const float* src = h + (long)tok * DIM;
    for (int j = threadIdx.x; j < DIM; j += 256) ein[dst + j] = src[j];
}

__global__ __launch_bounds__(256)
void moe_combine_k(const float* __restrict__ eout, const int* __restrict__ asg_e,
                   const int* __restrict__ asg_c, const float* __restrict__ asg_g,
                   float* x)
{
    const int tok = blockIdx.x;
    const int b = tok / SEQ;
    for (int kk = 0; kk < 2; ++kk) {
        int ai = kk * NTOK + tok;
        float gm = asg_g[ai];
        if (gm <= 0.f) continue;
        long src = ((long)asg_e[ai] * (BB * CAP) + b * CAP + asg_c[ai]) * DIM;
        for (int j = threadIdx.x; j < DIM; j += 256)
            x[(long)tok * DIM + j] += gm * eout[src + j];
    }
}

__global__ void write_aux_k(const float* __restrict__ auxacc, float* out, long idx)
{
    if (blockIdx.x == 0 && threadIdx.x == 0) out[idx] = auxacc[0];
}

// ---------------------------------------------------------------------------
// Host orchestration
// ---------------------------------------------------------------------------
static inline void launch_gemm(hipStream_t st, const float* A, const float* B,
                               const float* bias, const float* res, float* C,
                               int M, int N, int K, int lda, int ldb, int ldc,
                               int nbatch = 1, int nb2 = 1,
                               long sA1 = 0, long sA2 = 0, long sB1 = 0, long sB2 = 0,
                               long sC1 = 0, long sC2 = 0, long sBias = 0,
                               float alpha = 1.f, int transB = 0, int act = 0)
{
    dim3 grid((N + 127) / 128, (M + 127) / 128, nbatch);
    gemm_bf16_wmma<<<grid, 256, 0, st>>>(A, B, bias, res, C, M, N, K, lda, ldb, ldc,
                                         nb2, sA1, sA2, sB1, sB2, sC1, sC2, sBias,
                                         alpha, transB, act);
}

static void st_ff_host(hipStream_t st, float* mx, float* hh, float* t1, float* g,
                       const float* normg, const float* w1, const float* b1,
                       const float* mb, const float* w2, const float* b2)
{
    rmsnorm_k<<<NTOK, 256, 0, st>>>(mx, normg, hh);
    launch_gemm(st, hh, w1, b1, nullptr, t1, NTOK, 2 * MOEH, DIM, DIM, 2 * MOEH, 2 * MOEH);
    geglu_k<<<NTOK, 256, 0, st>>>(t1, mb, g, MOEH, NTOK);
    launch_gemm(st, g, w2, b2, mx, mx, NTOK, DIM, MOEH, MOEH, DIM, DIM);
}

extern "C" void kernel_launch(void* const* d_in, const int* in_sizes, int n_in,
                              void* d_out, int out_size, void* d_ws, size_t ws_size,
                              hipStream_t stream)
{
    (void)in_sizes; (void)n_in; (void)ws_size;
    const int*   ids   = (const int*)  d_in[0];
    const float* wte   = (const float*)d_in[2];
    const float* wpe   = (const float*)d_in[3];
    const float* ln1g  = (const float*)d_in[4];
    const float* ln1b  = (const float*)d_in[5];
    const float* ln2g  = (const float*)d_in[6];
    const float* ln2b  = (const float*)d_in[7];
    const float* aw    = (const float*)d_in[8];
    const float* ab    = (const float*)d_in[9];
    const float* apw   = (const float*)d_in[10];
    const float* apb   = (const float*)d_in[11];
    const float* mfw   = (const float*)d_in[12];
    const float* mfb   = (const float*)d_in[13];
    const float* mpw   = (const float*)d_in[14];
    const float* mpb   = (const float*)d_in[15];
    const float* lnfg  = (const float*)d_in[16];
    const float* lnfb  = (const float*)d_in[17];
    const float* ffbng = (const float*)d_in[18];
    const float* ffbw1 = (const float*)d_in[19];
    const float* ffbb1 = (const float*)d_in[20];
    const float* ffbmb = (const float*)d_in[21];
    const float* ffbw2 = (const float*)d_in[22];
    const float* ffbb2 = (const float*)d_in[23];
    const float* ffang = (const float*)d_in[24];
    const float* ffaw1 = (const float*)d_in[25];
    const float* ffab1 = (const float*)d_in[26];
    const float* ffamb = (const float*)d_in[27];
    const float* ffaw2 = (const float*)d_in[28];
    const float* ffab2 = (const float*)d_in[29];
    const float* moeng = (const float*)d_in[30];
    const float* gatew = (const float*)d_in[31];
    const float* ew1   = (const float*)d_in[32];
    const float* eb1   = (const float*)d_in[33];
    const float* emb   = (const float*)d_in[34];
    const float* ew2   = (const float*)d_in[35];
    const float* eb2   = (const float*)d_in[36];

    float* ws = (float*)d_ws;
    // ---- workspace layout (floats) ----
    float* x   = ws;                       // residual stream      786432
    float* h   = ws + 786432;              // LN out / moe stream  786432
    float* hh  = ws + 1572864;             // moe internal norm    786432
    float* scr = ws + 2359296;             // phase-overlapped     9830400
    // attention phase
    float* qkv = scr;                      // 1024*2304
    float* att = scr + 2359296;            // 2*12*512*512
    float* o   = scr + 8650752;            // 1024*768
    // mlp / st_ff phase
    float* t1  = scr;                      // up to 1024*4096
    float* gbuf= scr + 4194304;            // 1024*2048
    // moe expert phase
    float* ein  = scr;                     // 8*160*768   = 983040
    float* h1   = scr +  983040;           // 8*160*4096  = 5242880
    float* gg   = scr + 6225920;           // 8*160*2048  = 2621440
    float* eout = scr + 8847360;           // 8*160*768   = 983040
    // small buffers
    float* small  = ws + 12189696;
    float* rawsum = small;                 // 16
    float* zsum   = small + 16;            // 1
    float* auxacc = small + 17;            // 1
    int*   tok_e  = (int*)(small + 32);    // 2048
    float* tok_g  = small + 32 + 2048;     // 2048
    int*   asg_e  = (int*)(tok_g + 2048);  // 2048
    int*   asg_c  = asg_e + 2048;          // 2048
    float* asg_g  = (float*)(asg_c + 2048);// 2048

    const long NX = (long)NTOK * DIM;

    zero_k<<<1, 256, 0, stream>>>(auxacc, 1);
    embed_k<<<(int)((NX + 255) / 256), 256, 0, stream>>>(ids, wte, wpe, x);

    for (int i = 0; i < NLAY; ++i) {
        // ---- attention ----
        layernorm_k<<<NTOK, 256, 0, stream>>>(x, ln1g + i * DIM, ln1b + i * DIM, h);
        launch_gemm(stream, h, aw + (long)i * DIM * 3 * DIM, ab + (long)i * 3 * DIM,
                    nullptr, qkv, NTOK, 3 * DIM, DIM, DIM, 3 * DIM, 3 * DIM);
        // scores: q @ k^T / sqrt(HD), batched over (b, head)
        launch_gemm(stream, qkv, qkv + DIM, nullptr, nullptr, att,
                    SEQ, SEQ, HDIM, 3 * DIM, 3 * DIM, SEQ,
                    BB * NHEAD, NHEAD,
                    (long)SEQ * 3 * DIM, HDIM, (long)SEQ * 3 * DIM, HDIM,
                    (long)NHEAD * SEQ * SEQ, (long)SEQ * SEQ, 0,
                    0.125f, /*transB=*/1, 0);
        softmax_causal_k<<<BB * NHEAD * SEQ, 256, 0, stream>>>(att);
        // o = att @ v, written directly in (B,S,D) layout
        launch_gemm(stream, att, qkv + 2 * DIM, nullptr, nullptr, o,
                    SEQ, HDIM, SEQ, SEQ, 3 * DIM, DIM,
                    BB * NHEAD, NHEAD,
                    (long)NHEAD * SEQ * SEQ, (long)SEQ * SEQ,
                    (long)SEQ * 3 * DIM, HDIM,
                    (long)SEQ * DIM, HDIM, 0,
                    1.f, 0, 0);
        launch_gemm(stream, o, apw + (long)i * DIM * DIM, apb + (long)i * DIM,
                    x, x, NTOK, DIM, DIM, DIM, DIM, DIM);

        // ---- feed-forward ----
        layernorm_k<<<NTOK, 256, 0, stream>>>(x, ln2g + i * DIM, ln2b + i * DIM, h);
        if (i == 0 || i == 3) {
            // MoE block operates on h (mutated), then x += h
            st_ff_host(stream, h, hh, t1, gbuf, ffbng, ffbw1, ffbb1, ffbmb, ffbw2, ffbb2);
            rmsnorm_k<<<NTOK, 256, 0, stream>>>(h, moeng, hh);
            zero_k<<<1, 256, 0, stream>>>(rawsum, 17);
            gate_k<<<NTOK, 64, 0, stream>>>(hh, gatew, tok_e, tok_g, rawsum, zsum);
            moe_scan_k<<<1, 32, 0, stream>>>(tok_e, tok_g, asg_e, asg_c, asg_g,
                                             rawsum, zsum, auxacc, 1234u + (unsigned)i);
            zero_k<<<(983040 + 255) / 256, 256, 0, stream>>>(ein, 983040);
            moe_scatter_k<<<2 * NTOK, 256, 0, stream>>>(hh, asg_e, asg_c, asg_g, ein);
            launch_gemm(stream, ein, ew1, eb1, nullptr, h1,
                        BB * CAP, 2 * MOEH, DIM, DIM, 2 * MOEH, 2 * MOEH,
                        NE, 1, (long)BB * CAP * DIM, 0, (long)DIM * 2 * MOEH, 0,
                        (long)BB * CAP * 2 * MOEH, 0, 2 * MOEH);
            geglu_k<<<NE * BB * CAP, 256, 0, stream>>>(h1, emb, gg, MOEH, BB * CAP);
            launch_gemm(stream, gg, ew2, eb2, nullptr, eout,
                        BB * CAP, DIM, MOEH, MOEH, DIM, DIM,
                        NE, 1, (long)BB * CAP * MOEH, 0, (long)MOEH * DIM, 0,
                        (long)BB * CAP * DIM, 0, DIM);
            moe_combine_k<<<NTOK, 256, 0, stream>>>(eout, asg_e, asg_c, asg_g, h);
            st_ff_host(stream, h, hh, t1, gbuf, ffang, ffaw1, ffab1, ffamb, ffaw2, ffab2);
            add_k<<<(int)((NX + 255) / 256), 256, 0, stream>>>(x, h, NX);
        } else {
            launch_gemm(stream, h, mfw + (long)i * DIM * FFD, mfb + (long)i * FFD,
                        nullptr, t1, NTOK, FFD, DIM, DIM, FFD, FFD,
                        1, 1, 0, 0, 0, 0, 0, 0, 0, 1.f, 0, /*act=gelu_tanh*/1);
            launch_gemm(stream, t1, mpw + (long)i * FFD * DIM, mpb + (long)i * DIM,
                        x, x, NTOK, DIM, FFD, FFD, DIM, DIM);
        }
    }

    // ---- final LN + tied-embedding logits ----
    layernorm_k<<<NTOK, 256, 0, stream>>>(x, lnfg, lnfb, h);
    launch_gemm(stream, h, wte, nullptr, nullptr, (float*)d_out,
                NTOK, VOC, DIM, DIM, DIM, VOC,
                1, 1, 0, 0, 0, 0, 0, 0, 0, 1.f, /*transB=*/1, 0);
    write_aux_k<<<1, 32, 0, stream>>>(auxacc, (float*)d_out, (long)out_size - 1);
}
